// SwinTransformerBlock_73435350827031
// MI455X (gfx1250) — matrix operand order repaired and checked by reading, not compiled, not run on hardware
//
#include <hip/hip_runtime.h>
#include <hip/hip_bf16.h>
#include <math.h>

typedef __bf16 bf16_t;
typedef __attribute__((ext_vector_type(16))) __bf16 v16bf;
typedef __attribute__((ext_vector_type(8)))  __bf16 bf16x8;
typedef __attribute__((ext_vector_type(8)))  float  v8f;

static constexpr int SS_   = 3;
static constexpr int TOK   = 32 * 56 * 56;   // 100352 tokens
static constexpr int TOKP  = TOK + 64;       // +64 pad rows for 49->64 window padding

// ---------- small helpers ----------
__device__ __forceinline__ bf16_t f2bf(float f) {
    unsigned u = __builtin_bit_cast(unsigned, f);
    u += 0x7FFFu + ((u >> 16) & 1u);               // RNE
    unsigned short s = (unsigned short)(u >> 16);
    return __builtin_bit_cast(bf16_t, s);
}
__device__ __forceinline__ bf16_t bf0() {
    return __builtin_bit_cast(bf16_t, (unsigned short)0);
}
__device__ __forceinline__ v8f vzero() {
    v8f a;
#pragma unroll
    for (int i = 0; i < 8; ++i) a[i] = 0.0f;
    return a;
}
// build a v16bf WMMA fragment from two contiguous 16-byte runs (register concat)
__device__ __forceinline__ v16bf frag_from(const bf16_t* lo, const bf16_t* hi) {
    bf16x8 a = *(const bf16x8*)lo;
    bf16x8 b = *(const bf16x8*)hi;
    return __builtin_shufflevector(a, b, 0, 1, 2, 3, 4, 5, 6, 7,
                                         8, 9, 10, 11, 12, 13, 14, 15);
}

// ---------- prep: bf16 weights, combined bias+mask table, zero qkv pad rows ----------
static constexpr int WQKV = 768 * 256;
static constexpr int WPRJ = 256 * 256;
static constexpr int WFC1 = 1024 * 256;
static constexpr int WFC2 = 256 * 1024;
static constexpr int WTOT = WQKV + WPRJ + WFC1 + WFC2;   // 786432
static constexpr int NCMB = 4 * 8 * 64 * 64;              // 131072 combined table entries
static constexpr int NPAD = 64 * 768;                     // 49152

// label of token t inside a window of class (bh,bw); bh/bw = touches bottom/right edge
__device__ __forceinline__ int wlabel(int t, int bh, int bw) {
    int rb = bh ? ((t / 7) < 4 ? 1 : 2) : 0;
    int cb = bw ? ((t % 7) < 4 ? 1 : 2) : 0;
    return rb * 3 + cb;
}

__global__ void prep_kernel(const float* __restrict__ qkv_w, const float* __restrict__ proj_w,
                            const float* __restrict__ fc1_w, const float* __restrict__ fc2_w,
                            const float* __restrict__ rel_bias,
                            bf16_t* __restrict__ w_qkv, bf16_t* __restrict__ w_proj,
                            bf16_t* __restrict__ w_fc1, bf16_t* __restrict__ w_fc2,
                            float* __restrict__ combT, bf16_t* __restrict__ qkv_pad) {
    const int total = WTOT + NCMB + NPAD;
    for (int i = blockIdx.x * blockDim.x + threadIdx.x; i < total; i += gridDim.x * blockDim.x) {
        if (i < WQKV) {
            w_qkv[i] = f2bf(qkv_w[i]);
        } else if (i < WQKV + WPRJ) {
            int j = i - WQKV; w_proj[j] = f2bf(proj_w[j]);
        } else if (i < WQKV + WPRJ + WFC1) {
            int j = i - WQKV - WPRJ; w_fc1[j] = f2bf(fc1_w[j]);
        } else if (i < WTOT) {
            int j = i - WQKV - WPRJ - WFC1; w_fc2[j] = f2bf(fc2_w[j]);
        } else if (i < WTOT + NCMB) {
            int j = i - WTOT;                         // [cls][h][q][col]
            int col = j & 63, q = (j >> 6) & 63;
            int h = (j >> 12) & 7, cls = j >> 15;
            float v;
            if (col >= 49)      v = -1e30f;           // padding sentinel
            else if (q >= 49)   v = 0.0f;             // dead rows stay finite
            else {
                int ridx = (q / 7 - col / 7 + 6) * 13 + (q % 7 - col % 7 + 6);
                v = rel_bias[ridx * 8 + h];
                int bh = cls >> 1, bw = cls & 1;
                if (wlabel(q, bh, bw) != wlabel(col, bh, bw)) v -= 100.0f;
            }
            combT[j] = v;
        } else {
            qkv_pad[i - WTOT - NCMB] = bf0();
        }
    }
}

// ---------- layernorm (+optional shift/window-partition) -> bf16 rows ----------
__global__ __launch_bounds__(256) void ln_rows(const float* __restrict__ src,
                                               const float* __restrict__ gw,
                                               const float* __restrict__ gb,
                                               bf16_t* __restrict__ dst, int shifted) {
    const int row  = blockIdx.x * 8 + (threadIdx.x >> 5);
    const int lane = threadIdx.x & 31;
    if (row >= TOK) return;
    size_t srow = (size_t)row;
    if (shifted) {
        int b = row / 3136, rem = row % 3136;
        int widx = rem / 49, t = rem % 49;
        int wh = widx >> 3, ww = widx & 7;
        int i = wh * 7 + t / 7, j = ww * 7 + t % 7;
        int h = i + SS_; if (h >= 56) h -= 56;
        int w = j + SS_; if (w >= 56) w -= 56;
        srow = ((size_t)b * 56 + h) * 56 + w;
    }
    const float* xp = src + srow * 256;
    const int c0 = lane * 8;
    float4 v0 = *(const float4*)(xp + c0);
    float4 v1 = *(const float4*)(xp + c0 + 4);
    float x[8] = {v0.x, v0.y, v0.z, v0.w, v1.x, v1.y, v1.z, v1.w};
    float s = 0.f, ss = 0.f;
#pragma unroll
    for (int i = 0; i < 8; ++i) { s += x[i]; ss += x[i] * x[i]; }
#pragma unroll
    for (int off = 1; off < 32; off <<= 1) { s += __shfl_xor(s, off); ss += __shfl_xor(ss, off); }
    float mean = s * (1.f / 256.f);
    float var  = ss * (1.f / 256.f) - mean * mean;
    float rstd = rsqrtf(var + 1e-5f);
    float4 g0 = *(const float4*)(gw + c0), g1 = *(const float4*)(gw + c0 + 4);
    float4 b0 = *(const float4*)(gb + c0), b1 = *(const float4*)(gb + c0 + 4);
    float gg[8] = {g0.x, g0.y, g0.z, g0.w, g1.x, g1.y, g1.z, g1.w};
    float bb[8] = {b0.x, b0.y, b0.z, b0.w, b1.x, b1.y, b1.z, b1.w};
    bf16x8 o;
#pragma unroll
    for (int i = 0; i < 8; ++i) o[i] = f2bf((x[i] - mean) * rstd * gg[i] + bb[i]);
    *(bf16x8*)&dst[(size_t)row * 256 + c0] = o;
}

// ---------- generic bf16 WMMA GEMM: Y[M,N] = A[M,K] * W[N,K]^T + bias ----------
// MODE 0: bf16 store (QKV)    MODE 1: proj -> unshift scatter + residual (f32)
// MODE 2: GELU -> bf16 (FC1)  MODE 3: accumulate into f32 out (FC2 + residual)
template <int MODE, int KDIM, int NDIM>
__global__ __launch_bounds__(256) void gemm_bf16(const bf16_t* __restrict__ A,
                                                 const bf16_t* __restrict__ Wt,
                                                 const float* __restrict__ bias,
                                                 bf16_t* __restrict__ outb,
                                                 float* __restrict__ outf,
                                                 const float* __restrict__ resid) {
    __shared__ alignas(32) bf16_t As[128 * 32];
    __shared__ alignas(32) bf16_t Bs[128 * 32];
    const int tid = threadIdx.x;
    const int lane = tid & 31, wid = tid >> 5;
    const int half = lane >> 4, l16 = lane & 15;
    const int wm = wid & 3, wn = wid >> 2;          // 4 x 2 wave grid
    const int m0 = blockIdx.x * 128, n0 = blockIdx.y * 128;
    const int ldr = tid >> 1, ldc = (tid & 1) * 16; // 128 rows x 32 cols staging

    v8f acc[2][4];
#pragma unroll
    for (int a = 0; a < 2; ++a)
#pragma unroll
        for (int b = 0; b < 4; ++b) acc[a][b] = vzero();

    const bf16_t* ga = A  + (size_t)(m0 + ldr) * KDIM + ldc;
    const bf16_t* gb = Wt + (size_t)(n0 + ldr) * KDIM + ldc;

    for (int k0 = 0; k0 < KDIM; k0 += 32) {
        *(bf16x8*)&As[ldr * 32 + ldc]     = *(const bf16x8*)(ga + k0);
        *(bf16x8*)&As[ldr * 32 + ldc + 8] = *(const bf16x8*)(ga + k0 + 8);
        *(bf16x8*)&Bs[ldr * 32 + ldc]     = *(const bf16x8*)(gb + k0);
        *(bf16x8*)&Bs[ldr * 32 + ldc + 8] = *(const bf16x8*)(gb + k0 + 8);
        if (k0 + 32 < KDIM) {                        // prefetch next K tile into caches
            __builtin_prefetch((const void*)(ga + k0 + 32), 0, 3);
            __builtin_prefetch((const void*)(gb + k0 + 32), 0, 3);
        }
        __syncthreads();
        v16bf afr[2], bfr[4];
#pragma unroll
        for (int tm = 0; tm < 2; ++tm) {
            int r = wm * 32 + tm * 16 + l16;
            afr[tm] = frag_from(&As[r * 32 + half * 8], &As[r * 32 + 16 + half * 8]);
        }
#pragma unroll
        for (int tn = 0; tn < 4; ++tn) {
            int n = wn * 64 + tn * 16 + l16;
            bfr[tn] = frag_from(&Bs[n * 32 + half * 8], &Bs[n * 32 + 16 + half * 8]);
        }
#pragma unroll
        for (int tm = 0; tm < 2; ++tm)
#pragma unroll
            for (int tn = 0; tn < 4; ++tn)
                acc[tm][tn] = __builtin_amdgcn_wmma_f32_16x16x32_bf16(
                    false, afr[tm], false, bfr[tn], (short)0, acc[tm][tn], false, false);
        __syncthreads();
    }

#pragma unroll
    for (int tm = 0; tm < 2; ++tm) {
#pragma unroll
        for (int tn = 0; tn < 4; ++tn) {
            const int n = n0 + wn * 64 + tn * 16 + l16;
            const float bn = bias[n];
#pragma unroll
            for (int r = 0; r < 8; ++r) {
                const int m = m0 + wm * 32 + tm * 16 + r + half * 8;
                float y = acc[tm][tn][r] + bn;
                if (MODE == 0) {
                    outb[(size_t)m * NDIM + n] = f2bf(y);
                } else if (MODE == 2) {
                    float g = 0.5f * y * (1.0f + erff(y * 0.70710678118654752f));
                    outb[(size_t)m * NDIM + n] = f2bf(g);
                } else if (MODE == 3) {
                    outf[(size_t)m * NDIM + n] += y;
                } else {  // MODE 1: proj + un-window + reverse shift + residual
                    int win = m / 49, t = m % 49;
                    int b = win >> 6, widx = win & 63;
                    int wh = widx >> 3, ww = widx & 7;
                    int i = wh * 7 + t / 7, j = ww * 7 + t % 7;
                    int ho = i + SS_; if (ho >= 56) ho -= 56;
                    int wo = j + SS_; if (wo >= 56) wo -= 56;
                    size_t o = (((size_t)b * 56 + ho) * 56 + wo) * 256 + n;
                    outf[o] = resid[o] + y;
                }
            }
        }
    }
}

// ---------- attention: one wave per (window, head); N=49 padded to 64 ----------
__global__ __launch_bounds__(32) void attn_win(const bf16_t* __restrict__ qkv,   // [TOKP,768]
                                               const float* __restrict__ combT,  // [4,8,64,64]
                                               bf16_t* __restrict__ obuf) {      // [TOK,256]
    __shared__ alignas(32) bf16_t P[64 * 64];   // softmaxed probs, row-major [q][k]
    __shared__ alignas(32) bf16_t VT[32 * 64];  // v transposed [d][k], k padded with zeros
    const int wg = blockIdx.x;
    const int win = wg >> 3, head = wg & 7;
    const int lane = threadIdx.x;
    const int half = lane >> 4, l16 = lane & 15;
    const int widx = win & 63;
    const int wh = widx >> 3, ww = widx & 7;
    const int cls = ((wh == 7) ? 2 : 0) | ((ww == 7) ? 1 : 0);
    const float* Tp = combT + ((size_t)cls * 8 + head) * 4096;   // [64][64] slice

    const size_t base = (size_t)win * 49 * 768 + (size_t)head * 32;
    const bf16_t* qp = qkv + base;
    const bf16_t* kp = qkv + base + 256;
    const bf16_t* vp = qkv + base + 512;

    // stage V transposed in LDS (zero pad k>=49)
    for (int kk = lane; kk < 64; kk += 32) {
        if (kk < 49) {
            const bf16_t* vr = vp + (size_t)kk * 768;
#pragma unroll
            for (int d = 0; d < 32; ++d) VT[d * 64 + kk] = vr[d];
        } else {
#pragma unroll
            for (int d = 0; d < 32; ++d) VT[d * 64 + kk] = bf0();
        }
    }

    // preload K fragments (B-layout: lane holds column n=k-token, K=d in table order)
    v16bf kf[4];
#pragma unroll
    for (int tk = 0; tk < 4; ++tk) {
        const bf16_t* kr = kp + (size_t)(tk * 16 + l16) * 768 + half * 8;
        kf[tk] = frag_from(kr, kr + 16);
    }
    const float scale = 0.17677669529663687f;  // 1/sqrt(32)

    // phase 1: scores + softmax -> P (bias/mask/padding all folded into combT)
    for (int tq = 0; tq < 4; ++tq) {
        const bf16_t* qr = qp + (size_t)(tq * 16 + l16) * 768 + half * 8;
        v16bf af = frag_from(qr, qr + 16);
        v8f sc[4];
#pragma unroll
        for (int tk = 0; tk < 4; ++tk) sc[tk] = vzero();
#pragma unroll
        for (int tk = 0; tk < 4; ++tk)
            sc[tk] = __builtin_amdgcn_wmma_f32_16x16x32_bf16(
                false, af, false, kf[tk], (short)0, sc[tk], false, false);

#pragma unroll
        for (int r = 0; r < 8; ++r) {
            const int q = tq * 16 + r + half * 8;
            const float* Tr = Tp + q * 64 + l16;
            float vals[4];
#pragma unroll
            for (int tk = 0; tk < 4; ++tk) vals[tk] = fmaf(sc[tk][r], scale, Tr[tk * 16]);
            float m = fmaxf(fmaxf(vals[0], vals[1]), fmaxf(vals[2], vals[3]));
#pragma unroll
            for (int off = 1; off < 16; off <<= 1) m = fmaxf(m, __shfl_xor(m, off));
            float e[4], sum = 0.f;
#pragma unroll
            for (int tk = 0; tk < 4; ++tk) { e[tk] = __expf(vals[tk] - m); sum += e[tk]; }
#pragma unroll
            for (int off = 1; off < 16; off <<= 1) sum += __shfl_xor(sum, off);
            float inv = 1.0f / sum;
#pragma unroll
            for (int tk = 0; tk < 4; ++tk) P[q * 64 + tk * 16 + l16] = f2bf(e[tk] * inv);
        }
    }
    __syncthreads();

    // phase 2: O = P @ V
    for (int tq = 0; tq < 4; ++tq) {
        v8f oa[2];
        oa[0] = vzero(); oa[1] = vzero();
#pragma unroll
        for (int kk = 0; kk < 64; kk += 32) {
            const bf16_t* pr = &P[(tq * 16 + l16) * 64 + kk + half * 8];
            v16bf af = frag_from(pr, pr + 16);
#pragma unroll
            for (int td = 0; td < 2; ++td) {
                const bf16_t* br = &VT[(td * 16 + l16) * 64 + kk + half * 8];
                v16bf bfrag = frag_from(br, br + 16);
                oa[td] = __builtin_amdgcn_wmma_f32_16x16x32_bf16(
                    false, af, false, bfrag, (short)0, oa[td], false, false);
            }
        }
#pragma unroll
        for (int td = 0; td < 2; ++td) {
#pragma unroll
            for (int r = 0; r < 8; ++r) {
                int q = tq * 16 + r + half * 8;
                if (q < 49) {
                    int d = td * 16 + l16;
                    obuf[((size_t)win * 49 + q) * 256 + head * 32 + d] = f2bf(oa[td][r]);
                }
            }
        }
    }
}

// ---------- launcher ----------
extern "C" void kernel_launch(void* const* d_in, const int* in_sizes, int n_in,
                              void* d_out, int out_size, void* d_ws, size_t ws_size,
                              hipStream_t stream) {
    const float* x       = (const float*)d_in[0];
    const float* norm1_w = (const float*)d_in[1];
    const float* norm1_b = (const float*)d_in[2];
    const float* qkv_w   = (const float*)d_in[3];
    const float* qkv_b   = (const float*)d_in[4];
    const float* proj_w  = (const float*)d_in[5];
    const float* proj_b  = (const float*)d_in[6];
    const float* rel_bias= (const float*)d_in[7];
    const float* norm2_w = (const float*)d_in[8];
    const float* norm2_b = (const float*)d_in[9];
    const float* fc1_w   = (const float*)d_in[10];
    const float* fc1_b   = (const float*)d_in[11];
    const float* fc2_w   = (const float*)d_in[12];
    const float* fc2_b   = (const float*)d_in[13];
    float* out = (float*)d_out;

    char* ws = (char*)d_ws;
    bf16_t* w_qkv  = (bf16_t*)(ws + 0);
    bf16_t* w_proj = (bf16_t*)(ws + 393216);
    bf16_t* w_fc1  = (bf16_t*)(ws + 524288);
    bf16_t* w_fc2  = (bf16_t*)(ws + 1048576);
    float*  combT  = (float*) (ws + 1572864);                   // 512 KB combined table
    size_t off_win = 2097152;
    bf16_t* winb   = (bf16_t*)(ws + off_win);                   // [TOKP,256] bf16
    size_t off_qkv = off_win + (size_t)TOKP * 256 * 2;
    bf16_t* qkvb   = (bf16_t*)(ws + off_qkv);                   // [TOKP,768] bf16
    size_t off_obu = off_qkv + (size_t)TOKP * 768 * 2;
    bf16_t* obuf   = (bf16_t*)(ws + off_obu);                   // [TOK,256] bf16
    bf16_t* h_in   = winb;                                      // reuse (win dead after QKV)
    bf16_t* h_mid  = qkvb;                                      // reuse (qkv+obuf dead after proj)
    bf16_t* qkv_pad= qkvb + (size_t)TOK * 768;

    // 1) weight conversion + combined bias/mask table + qkv pad zeroing
    prep_kernel<<<3776, 256, 0, stream>>>(qkv_w, proj_w, fc1_w, fc2_w, rel_bias,
                                          w_qkv, w_proj, w_fc1, w_fc2, combT, qkv_pad);
    // 2) LN1 + shift + window partition -> bf16
    ln_rows<<<TOK / 8, 256, 0, stream>>>(x, norm1_w, norm1_b, winb, 1);
    // 3) QKV GEMM: [100352,256] x [768,256]^T
    gemm_bf16<0, 256, 768><<<dim3(784, 6), 256, 0, stream>>>(winb, w_qkv, qkv_b, qkvb, nullptr, nullptr);
    // 4) windowed attention (2048 windows x 8 heads)
    attn_win<<<16384, 32, 0, stream>>>(qkvb, combT, obuf);
    // 5) proj GEMM + un-window + reverse shift + residual -> d_out (= x1)
    gemm_bf16<1, 256, 256><<<dim3(784, 2), 256, 0, stream>>>(obuf, w_proj, proj_b, nullptr, out, x);
    // 6) LN2 -> bf16
    ln_rows<<<TOK / 8, 256, 0, stream>>>(out, norm2_w, norm2_b, h_in, 0);
    // 7) FC1 + GELU: [100352,256] x [1024,256]^T
    gemm_bf16<2, 256, 1024><<<dim3(784, 8), 256, 0, stream>>>(h_in, w_fc1, fc1_b, h_mid, nullptr, nullptr);
    // 8) FC2 + residual accumulate into d_out
    gemm_bf16<3, 1024, 256><<<dim3(784, 2), 256, 0, stream>>>(h_mid, w_fc2, fc2_b, nullptr, out, nullptr);
}
